// BoneLinear_22600117911584
// MI455X (gfx1250) — matrix operand order repaired and checked by reading, not compiled, not run on hardware
//
#include <hip/hip_runtime.h>
#include <hip/hip_bf16.h>

typedef __attribute__((ext_vector_type(16))) _Float16 v16h;
typedef __attribute__((ext_vector_type(4)))  _Float16 v4h;
typedef __attribute__((ext_vector_type(8)))  float    v8f;

#define K_DIM 4096   // IN features
#define N_DIM 4096   // OUT features

// ---------------------------------------------------------------------------
// Kernel 1: build effective weight (f16) into workspace.
//   W_eff[a*64+i, b*64+k] = Wblk[i,k] + sum_j Wblk[i,j]*bone[b,j,k] + bone[b,i,k]
// ---------------------------------------------------------------------------
__global__ __launch_bounds__(256)
void bone_weff_kernel(const float* __restrict__ weight,
                      const float* __restrict__ bone,
                      _Float16* __restrict__ weff) {
    __shared__ __align__(16) float sW[64 * 64];
    __shared__ __align__(16) float sBn[64 * 64];
    const int bb  = blockIdx.x;   // b (K block index)
    const int aa  = blockIdx.y;   // a (OUT block index)
    const int tid = threadIdx.x;

    #pragma unroll
    for (int p = 0; p < 4; ++p) {
        int lin = p * 1024 + tid * 4;
        int i = lin >> 6;
        int j = lin & 63;
        *(float4*)&sW[lin]  = *(const float4*)&weight[(size_t)(aa * 64 + i) * K_DIM + bb * 64 + j];
        *(float4*)&sBn[lin] = *(const float4*)&bone[(size_t)bb * 4096 + lin];
    }
    __syncthreads();

    const int i  = tid >> 2;
    const int kb = (tid & 3) * 16;

    float acc[16];
    #pragma unroll
    for (int kk = 0; kk < 16; ++kk) acc[kk] = 0.0f;

    for (int j = 0; j < 64; ++j) {
        float wv = sW[i * 64 + j];
        #pragma unroll
        for (int kk = 0; kk < 16; ++kk)
            acc[kk] = fmaf(wv, sBn[j * 64 + kb + kk], acc[kk]);
    }

    #pragma unroll
    for (int kk = 0; kk < 16; ++kk) {
        float r = sW[i * 64 + kb + kk] + acc[kk] + sBn[i * 64 + kb + kk];
        weff[(size_t)(aa * 64 + i) * K_DIM + bb * 64 + kb + kk] = (_Float16)r;
    }
}

// ---------------------------------------------------------------------------
// Kernel 2: y = x @ W_eff^T with v_wmma_f32_16x16x32_f16.
//   Block tile 128(M) x 256(N), K-step 32, 8 waves; wave tile 64x64 = 4x4
//   WMMA fragments -> 16 WMMAs per K-step per wave, 1:1 wmma:ds_load ratio.
//   B tile staged with GLOBAL_LOAD_ASYNC_TO_LDS_B128 (ASYNCcnt path).
// ---------------------------------------------------------------------------
__global__ __launch_bounds__(256)
void bone_gemm_kernel(const float* __restrict__ x,
                      const _Float16* __restrict__ weff,
                      float* __restrict__ out) {
    __shared__ __align__(16) _Float16 sA[128 * 32];  // x tile [m][k] f16, 8 KB
    __shared__ __align__(16) _Float16 sB[256 * 32];  // W_eff tile [n][k] f16, 16 KB

    const int tid   = threadIdx.x;
    const int lane  = tid & 31;
    const int wave  = tid >> 5;
    const int wm    = wave >> 2;            // 0..1  (64-row M sub-panel)
    const int wn    = wave & 3;             // 0..3  (64-col N sub-panel)
    const int mBase = blockIdx.y * 128;
    const int nBase = blockIdx.x * 256;
    const int l16   = lane & 15;
    const int lh    = lane >> 4;            // lane half (K-group select)

    v8f acc[4][4];
    #pragma unroll
    for (int i = 0; i < 4; ++i)
        #pragma unroll
        for (int j = 0; j < 4; ++j) {
            v8f z = {};
            acc[i][j] = z;
        }

    // per-thread staging coordinates
    const int arow = tid >> 3;              // A: base row (32-row strides)
    const int acol = (tid & 7) * 4;         // A: 4 fp32 columns
    const int brow = tid >> 2;              // B: base row (64-row strides)
    const int bcol = (tid & 3) * 8;         // B: 8 f16 columns (16 bytes)

    for (int k0 = 0; k0 < K_DIM; k0 += 32) {
        // ---- stage A: 128x32 fp32 -> f16 in LDS (needs conversion: VGPR path) ----
        #pragma unroll
        for (int p = 0; p < 4; ++p) {
            int row = p * 32 + arow;
            float4 v = *(const float4*)&x[(size_t)(mBase + row) * K_DIM + k0 + acol];
            v4h h;
            h.x = (_Float16)v.x; h.y = (_Float16)v.y;
            h.z = (_Float16)v.z; h.w = (_Float16)v.w;
            *(v4h*)&sA[row * 32 + acol] = h;
        }
        // ---- stage B: 256x32 f16 pure copy -> async global->LDS DMA ----
        #pragma unroll
        for (int p = 0; p < 4; ++p) {
            int row = p * 64 + brow;
            unsigned int      ldsaddr = (unsigned int)(uintptr_t)&sB[row * 32 + bcol];
            unsigned long long gaddr  =
                (unsigned long long)(uintptr_t)&weff[(size_t)(nBase + row) * K_DIM + k0 + bcol];
            asm volatile("global_load_async_to_lds_b128 %0, %1, off"
                         :: "v"(ldsaddr), "v"(gaddr) : "memory");
        }
        if (k0 + 32 < K_DIM)
            __builtin_prefetch(&x[(size_t)(mBase + arow) * K_DIM + k0 + 32], 0, 1);

        asm volatile("s_wait_asynccnt 0x0" ::: "memory");
        __syncthreads();

        // ---- B fragments: lane = (N = l16), K = lh*8+0..7 and 16+lh*8+0..7 ----
        v16h bfrag[4];
        #pragma unroll
        for (int j = 0; j < 4; ++j) {
            int n = wn * 64 + j * 16 + l16;
            ((float4*)&bfrag[j])[0] = *(const float4*)&sB[n * 32 + lh * 8];
            ((float4*)&bfrag[j])[1] = *(const float4*)&sB[n * 32 + 16 + lh * 8];
        }

        // ---- A fragments + 16 WMMAs ----
        #pragma unroll
        for (int i = 0; i < 4; ++i) {
            int rowm = wm * 64 + i * 16 + l16;   // M = lane%16
            v16h afrag;
            // A layout: VGPR0-3 hold K = lh*8+0..7, VGPR4-7 hold K = 16+lh*8+0..7
            ((float4*)&afrag)[0] = *(const float4*)&sA[rowm * 32 + lh * 8];
            ((float4*)&afrag)[1] = *(const float4*)&sA[rowm * 32 + 16 + lh * 8];
            #pragma unroll
            for (int j = 0; j < 4; ++j)
                acc[i][j] = __builtin_amdgcn_wmma_f32_16x16x32_f16(
                    /*neg_a=*/false, afrag, /*neg_b=*/false, bfrag[j],
                    /*c_mod=*/(short)0, acc[i][j],
                    /*reuse_a=*/false, /*reuse_b=*/false);
        }
        __syncthreads();
    }

    // ---- epilogue: C/D layout -> VGPR v: M = v + 8*(lane/16), N = lane%16 ----
    #pragma unroll
    for (int i = 0; i < 4; ++i)
        #pragma unroll
        for (int j = 0; j < 4; ++j) {
            int n    = nBase + wn * 64 + j * 16 + l16;
            int mrow = mBase + wm * 64 + i * 16 + lh * 8;
            #pragma unroll
            for (int v = 0; v < 8; ++v)
                out[(size_t)(mrow + v) * N_DIM + n] = acc[i][j][v];
        }
}

extern "C" void kernel_launch(void* const* d_in, const int* in_sizes, int n_in,
                              void* d_out, int out_size, void* d_ws, size_t ws_size,
                              hipStream_t stream) {
    const float* x      = (const float*)d_in[0];   // (B,T,IN) fp32
    const float* weight = (const float*)d_in[1];   // (OUT,IN) fp32
    const float* bone   = (const float*)d_in[2];   // (IN/R,R,R) fp32
    float* out          = (float*)d_out;           // (B,T,OUT) fp32
    _Float16* weff      = (_Float16*)d_ws;         // 4096*4096 f16 = 32 MB scratch

    const int M = in_sizes[0] / K_DIM;             // B*T = 16384

    bone_weff_kernel<<<dim3(K_DIM / 64, N_DIM / 64), 256, 0, stream>>>(weight, bone, weff);
    bone_gemm_kernel<<<dim3(N_DIM / 256, M / 128), 256, 0, stream>>>(x, weff, out);
}
